// SelfAttention_3994319586026
// MI455X (gfx1250) — matrix operand order repaired and checked by reading, not compiled, not run on hardware
//
#include <hip/hip_runtime.h>
#include <hip/hip_bf16.h>

// ---------------- problem constants ----------------
constexpr int BATCH = 4;
constexpr int SEQ   = 2048;
constexpr int DIM   = 1024;

typedef __bf16 bf16_t;
typedef __attribute__((ext_vector_type(16))) __bf16 v16bf;
typedef __attribute__((ext_vector_type(8)))  __bf16 v8bf;
typedef __attribute__((ext_vector_type(8)))  float  v8f;
typedef __attribute__((ext_vector_type(4)))  float  v4f;

// ---------------- workspace layout (bytes) ----------------
constexpr size_t QKV_ELEMS = (size_t)BATCH * SEQ * DIM;     // 8,388,608
constexpr size_t BY_QKV    = QKV_ELEMS * 2;                 // 16 MB each
constexpr size_t OFF_Q     = 0;
constexpr size_t OFF_K     = BY_QKV;
constexpr size_t OFF_V     = 2 * BY_QKV;                    // V stored transposed [B][D][S]
constexpr size_t OFF_W     = 3 * BY_QKV;                    // 3 transposed bf16 weights
constexpr size_t BY_W      = (size_t)3 * DIM * DIM * 2;     // 6 MB
constexpr size_t OFF_P     = OFF_W + BY_W;                  // P bf16 [B][S][S]
constexpr size_t BY_P      = (size_t)BATCH * SEQ * SEQ * 2; // 32 MB
constexpr size_t OFF_L     = OFF_P + BY_P;                  // row sums f32 [B*S]

// ---------------- helpers ----------------
__device__ inline v8f wmma_bf16(v16bf a, v16bf b, v8f c) {
  return __builtin_amdgcn_wmma_f32_16x16x32_bf16(
      /*neg_a=*/false, a, /*neg_b=*/false, b,
      /*c_mod=*/(short)0, c, /*reuse_a=*/false, /*reuse_b=*/false);
}

__device__ inline v8bf zero_v8bf() {
  v8bf z;
#pragma unroll
  for (int i = 0; i < 8; ++i) z[i] = (__bf16)0.0f;
  return z;
}

__device__ inline v16bf cat16(v8bf lo, v8bf hi) {
  return __builtin_shufflevector(lo, hi, 0, 1, 2, 3, 4, 5, 6, 7,
                                 8, 9, 10, 11, 12, 13, 14, 15);
}

// A-operand (16x32 bf16, MxK) loaded from f32 row-major src, converting.
// lane<16: M=lane, K = {0..7, 16..23}; lane>=16: M=lane-16, K = {8..15, 24..31}
__device__ inline v16bf load_A_from_f32(const float* src, int ld, int m0, int kk) {
  const int lane = threadIdx.x & 31;
  const int m    = m0 + (lane & 15);
  const int k0   = kk + ((lane >> 4) << 3);     // 0 or 8
  const v4f* p   = (const v4f*)(src + (size_t)m * ld + k0);
  v4f f0 = p[0], f1 = p[1];                     // K = k0 .. k0+7
  v4f f2 = p[4], f3 = p[5];                     // K = k0+16 .. k0+23
  v16bf a;
#pragma unroll
  for (int i = 0; i < 4; ++i) {
    a[i]      = (__bf16)f0[i];
    a[4 + i]  = (__bf16)f1[i];
    a[8 + i]  = (__bf16)f2[i];
    a[12 + i] = (__bf16)f3[i];
  }
  return a;
}

// A-operand (16x32 bf16) from bf16 row-major src.
__device__ inline v16bf load_A_bf16(const bf16_t* src, int ld, int m0, int kk) {
  const int lane = threadIdx.x & 31;
  const int m    = m0 + (lane & 15);
  const int k0   = kk + ((lane >> 4) << 3);     // 0 or 8
  const v8bf* p  = (const v8bf*)(src + (size_t)m * ld + k0);
  return cat16(p[0], p[2]);                     // +16 elements == +2 v8bf
}

// B-operand (32x16 bf16, KxN) from "N-major" bf16 src (row n holds K contiguous).
// lane<16: N=lane, K = {0..7, 16..23}; lane>=16: N=lane-16, K = {8..15, 24..31}
__device__ inline v16bf load_B_bf16(const bf16_t* src, int ld, int n0, int kk) {
  const int lane = threadIdx.x & 31;
  const int n    = n0 + (lane & 15);
  const int k0   = kk + ((lane >> 4) << 3);     // 0 or 8
  const v8bf* p  = (const v8bf*)(src + (size_t)n * ld + k0);
  return cat16(p[0], p[2]);
}

// ---------------- kernel 1: W -> W^T bf16 (3 weights) ----------------
__global__ void wcvt_kernel(const float* __restrict__ WQ,
                            const float* __restrict__ WK,
                            const float* __restrict__ WV,
                            bf16_t* __restrict__ Wt3) {
  const size_t idx = (size_t)blockIdx.x * blockDim.x + threadIdx.x; // over 3*D*D
  const int which = (int)(idx / ((size_t)DIM * DIM));
  const int r     = (int)(idx % ((size_t)DIM * DIM));
  const int n = r / DIM;          // output row (contiguous writes over k)
  const int k = r % DIM;
  const float* W = (which == 0) ? WQ : ((which == 1) ? WK : WV);
  Wt3[(size_t)which * DIM * DIM + (size_t)n * DIM + k] =
      (__bf16)W[(size_t)k * DIM + n];
}

// ---------------- kernel 2: Q/K/V projection GEMM (bf16 WMMA) ----------------
// grid: (B*S/32, D/64, 3), block: 32. One wave -> 32x64 tile (2x4 WMMA tiles).
// Per K-step: 2 A-loads + 4 B-loads feed 8 WMMAs.
__global__ __launch_bounds__(32) void qkv_proj_kernel(
    const float* __restrict__ x, const bf16_t* __restrict__ Wt3,
    bf16_t* __restrict__ Qb, bf16_t* __restrict__ Kb, bf16_t* __restrict__ VbT) {
  const int m0    = blockIdx.x * 32;            // row in [0, B*S)
  const int n0    = blockIdx.y * 64;            // col in [0, D)
  const int which = blockIdx.z;
  const bf16_t* Wt = Wt3 + (size_t)which * DIM * DIM;

  v8f acc[2][4];
#pragma unroll
  for (int mb = 0; mb < 2; ++mb)
#pragma unroll
    for (int nb = 0; nb < 4; ++nb)
#pragma unroll
      for (int i = 0; i < 8; ++i) acc[mb][nb][i] = 0.0f;

  for (int kk = 0; kk < DIM; kk += 32) {
    v16bf a0 = load_A_from_f32(x, DIM, m0, kk);
    v16bf a1 = load_A_from_f32(x, DIM, m0 + 16, kk);
#pragma unroll
    for (int nb = 0; nb < 4; ++nb) {
      v16bf b = load_B_bf16(Wt, DIM, n0 + nb * 16, kk);
      acc[0][nb] = wmma_bf16(a0, b, acc[0][nb]);
      acc[1][nb] = wmma_bf16(a1, b, acc[1][nb]);
    }
  }

  const int lane = threadIdx.x & 31;
  const int half = lane >> 4;
  if (which == 2) {
    // V stored transposed: VbT[b][d][t]
    const int b = m0 / SEQ;
#pragma unroll
    for (int mb = 0; mb < 2; ++mb) {
      const int t0 = (m0 % SEQ) + mb * 16 + 8 * half;
#pragma unroll
      for (int nb = 0; nb < 4; ++nb) {
        const int nn = n0 + nb * 16 + (lane & 15);
        v8bf pk;
#pragma unroll
        for (int r = 0; r < 8; ++r) pk[r] = (__bf16)acc[mb][nb][r];
        *(v8bf*)(VbT + (size_t)b * DIM * SEQ + (size_t)nn * SEQ + t0) = pk;
      }
    }
  } else {
    bf16_t* out = (which == 0) ? Qb : Kb;
#pragma unroll
    for (int mb = 0; mb < 2; ++mb) {
      const int mbase = m0 + mb * 16 + 8 * half;
#pragma unroll
      for (int nb = 0; nb < 4; ++nb) {
        const int nn = n0 + nb * 16 + (lane & 15);
#pragma unroll
        for (int r = 0; r < 8; ++r)
          out[(size_t)(mbase + r) * DIM + nn] = (__bf16)acc[mb][nb][r];
      }
    }
  }
}

// ---------------- kernel 3: causal softmax, P = exp(s - m), rowsum l ----------
// grid: (S/16, B), block: 32. Two passes over key tiles (recompute scores).
// 4 key tiles per outer step share each Q A-load; causal tail handled by the
// mask (rounded-up tiles read valid K rows and produce fully-masked columns).
__global__ __launch_bounds__(32) void softmax_p_kernel(
    const bf16_t* __restrict__ Qb, const bf16_t* __restrict__ Kb,
    bf16_t* __restrict__ P, float* __restrict__ Lsum) {
  const int qt = blockIdx.x;
  const int b  = blockIdx.y;
  const int q0 = qt * 16;
  const bf16_t* Q  = Qb + (size_t)b * SEQ * DIM;
  const bf16_t* K  = Kb + (size_t)b * SEQ * DIM;
  bf16_t*       Pb = P + (size_t)b * SEQ * SEQ;

  const int lane  = threadIdx.x & 31;
  const int half  = lane >> 4;
  const int nlane = lane & 15;
  const float scale = 0.03125f;                 // 1/sqrt(1024)
  const int njr = ((qt + 1) + 3) & ~3;          // causal tiles rounded to 4

  float m[8];
#pragma unroll
  for (int r = 0; r < 8; ++r) m[r] = -__builtin_inff();

  // ---- pass 1: row maxima ----
  for (int j = 0; j < njr; j += 4) {
    v8f c[4];
#pragma unroll
    for (int jb = 0; jb < 4; ++jb)
#pragma unroll
      for (int i = 0; i < 8; ++i) c[jb][i] = 0.0f;
    for (int kk = 0; kk < DIM; kk += 32) {
      v16bf a = load_A_bf16(Q, DIM, q0, kk);
#pragma unroll
      for (int jb = 0; jb < 4; ++jb)
        c[jb] = wmma_bf16(a, load_B_bf16(K, DIM, (j + jb) * 16, kk), c[jb]);
    }
#pragma unroll
    for (int jb = 0; jb < 4; ++jb) {
#pragma unroll
      for (int r = 0; r < 8; ++r) {
        const int qi = q0 + r + 8 * half;
        const int kj = (j + jb) * 16 + nlane;
        float s = (kj <= qi) ? c[jb][r] * scale : -__builtin_inff();
        s = fmaxf(s, __shfl_xor(s, 1));
        s = fmaxf(s, __shfl_xor(s, 2));
        s = fmaxf(s, __shfl_xor(s, 4));
        s = fmaxf(s, __shfl_xor(s, 8));
        m[r] = fmaxf(m[r], s);
      }
    }
  }

  // ---- pass 2: P = exp(s - m), accumulate row sums ----
  float l[8];
#pragma unroll
  for (int r = 0; r < 8; ++r) l[r] = 0.0f;

  for (int j = 0; j < njr; j += 4) {
    v8f c[4];
#pragma unroll
    for (int jb = 0; jb < 4; ++jb)
#pragma unroll
      for (int i = 0; i < 8; ++i) c[jb][i] = 0.0f;
    for (int kk = 0; kk < DIM; kk += 32) {
      v16bf a = load_A_bf16(Q, DIM, q0, kk);
#pragma unroll
      for (int jb = 0; jb < 4; ++jb)
        c[jb] = wmma_bf16(a, load_B_bf16(K, DIM, (j + jb) * 16, kk), c[jb]);
    }
#pragma unroll
    for (int jb = 0; jb < 4; ++jb) {
#pragma unroll
      for (int r = 0; r < 8; ++r) {
        const int qi = q0 + r + 8 * half;
        const int kj = (j + jb) * 16 + nlane;
        const float s = c[jb][r] * scale;
        const float p = (kj <= qi) ? __expf(s - m[r]) : 0.0f;
        l[r] += p;
        Pb[(size_t)qi * SEQ + kj] = (__bf16)p;
      }
    }
  }

#pragma unroll
  for (int r = 0; r < 8; ++r) {
    l[r] += __shfl_xor(l[r], 1);
    l[r] += __shfl_xor(l[r], 2);
    l[r] += __shfl_xor(l[r], 4);
    l[r] += __shfl_xor(l[r], 8);
  }
  if (nlane == 0) {
#pragma unroll
    for (int r = 0; r < 8; ++r)
      Lsum[(size_t)b * SEQ + q0 + r + 8 * half] = l[r];
  }
}

// ---------------- kernel 4: O = (P @ V) / l  (bf16 WMMA, f32 out) -----------
// grid: (S/16, D/64, B), block: 32. 4 d-tiles per wave share each P A-load.
__global__ __launch_bounds__(32) void pv_kernel(
    const bf16_t* __restrict__ P, const bf16_t* __restrict__ VbT,
    const float* __restrict__ Lsum, float* __restrict__ out) {
  const int qt = blockIdx.x;
  const int d0 = blockIdx.y * 64;
  const int b  = blockIdx.z;
  const int q0 = qt * 16;
  const bf16_t* Pb = P + (size_t)b * SEQ * SEQ;
  const bf16_t* Vt = VbT + (size_t)b * DIM * SEQ;
  const int nkeys = (qt + 1) * 16;

  const int lane  = threadIdx.x & 31;
  const int half  = lane >> 4;
  const int nlane = lane & 15;

  v8f c[4];
#pragma unroll
  for (int nb = 0; nb < 4; ++nb)
#pragma unroll
    for (int i = 0; i < 8; ++i) c[nb][i] = 0.0f;

  for (int k = 0; k < nkeys; k += 32) {
    // A = P[q0..q0+15][k..k+31]; zero-mask chunks beyond causal frontier.
    const int mrow = q0 + nlane;
    const int k0   = k + (half << 3);
    const v8bf* p0 = (const v8bf*)(Pb + (size_t)mrow * SEQ + k0);
    v8bf lo = (k0 < nkeys) ? p0[0] : zero_v8bf();
    v8bf hi = (k0 + 16 < nkeys) ? p0[2] : zero_v8bf();
    v16bf a = cat16(lo, hi);
    // B = V^T rows d0.., keys k..k+31 (contiguous per lane)
#pragma unroll
    for (int nb = 0; nb < 4; ++nb)
      c[nb] = wmma_bf16(a, load_B_bf16(Vt, SEQ, d0 + nb * 16, k), c[nb]);
  }

#pragma unroll
  for (int r = 0; r < 8; ++r) {
    const int qi = q0 + r + 8 * half;
    const float linv = 1.0f / Lsum[(size_t)b * SEQ + qi];
#pragma unroll
    for (int nb = 0; nb < 4; ++nb)
      out[((size_t)b * SEQ + qi) * DIM + d0 + nb * 16 + nlane] = c[nb][r] * linv;
  }
}

// ---------------- launch ----------------
extern "C" void kernel_launch(void* const* d_in, const int* in_sizes, int n_in,
                              void* d_out, int out_size, void* d_ws, size_t ws_size,
                              hipStream_t stream) {
  const float* x  = (const float*)d_in[0];
  const float* WQ = (const float*)d_in[1];
  const float* WK = (const float*)d_in[2];
  const float* WV = (const float*)d_in[3];
  float* out = (float*)d_out;

  char* ws = (char*)d_ws;
  bf16_t* Qb  = (bf16_t*)(ws + OFF_Q);
  bf16_t* Kb  = (bf16_t*)(ws + OFF_K);
  bf16_t* VbT = (bf16_t*)(ws + OFF_V);
  bf16_t* Wt3 = (bf16_t*)(ws + OFF_W);
  bf16_t* Pb  = (bf16_t*)(ws + OFF_P);
  float*  Ls  = (float*)(ws + OFF_L);

  // 1) weights -> transposed bf16
  {
    const size_t total = (size_t)3 * DIM * DIM;
    wcvt_kernel<<<dim3((unsigned)(total / 256)), 256, 0, stream>>>(WQ, WK, WV, Wt3);
  }
  // 2) Q, K, V projections (32x64 tile per wave)
  qkv_proj_kernel<<<dim3(BATCH * SEQ / 32, DIM / 64, 3), 32, 0, stream>>>(
      x, Wt3, Qb, Kb, VbT);
  // 3) causal softmax -> P (bf16), row sums
  softmax_p_kernel<<<dim3(SEQ / 16, BATCH), 32, 0, stream>>>(Qb, Kb, Pb, Ls);
  // 4) O = (P @ V) / l (16x64 tile per wave)
  pv_kernel<<<dim3(SEQ / 16, DIM / 64, BATCH), 32, 0, stream>>>(Pb, VbT, Ls, out);
}
